// SpatialEncoder_89541478187112
// MI455X (gfx1250) — matrix operand order repaired and checked by reading, not compiled, not run on hardware
//
#include <hip/hip_runtime.h>
#include <hip/hip_bf16.h>

typedef __attribute__((ext_vector_type(16))) _Float16 v16h;
typedef __attribute__((ext_vector_type(8)))  _Float16 v8h;
typedef __attribute__((ext_vector_type(8)))  float    v8f;

#define NN  7
#define HH  128
#define FIN 32
#define TOK_PER_WAVE  16
#define WAVES_PER_BLK 4
#define TOK_PER_BLK   (TOK_PER_WAVE * WAVES_PER_BLK)   // 64
#define ROWS          (TOK_PER_WAVE * NN)              // 112
#define BT            16384

// ---- LDS layout (bytes) ----
#define OFF_W    0         // 28672 halves (Wp^T 896x32; reused for W^T 128x128)
#define OFF_H    57344     // 4 waves * 14336 halves  (h,  [112][128] f16 per wave)
#define OFF_WH   172032    // 4 waves * 14336 halves  (Wh, [112][128] f16 per wave)
#define OFF_FS   286720    // 4 * 112 f32
#define OFF_FD   288512    // 4 * 112 f32
#define OFF_AV   290304    // 256 f32 (attention vector a)
#define OFF_G    291328    // 128 f32
#define OFF_B    291840    // 128 f32
#define OFF_ADJ  292352    // 49 f32 (padded)
#define SMEM_BYTES 292608

__device__ __forceinline__ v16h cat8(v8h lo, v8h hi) {
  return __builtin_shufflevector(lo, hi, 0,1,2,3,4,5,6,7,8,9,10,11,12,13,14,15);
}

// A fragment 16x32 f16 from LDS row-major [.][>=32] tile.
// Per ISA 7.12.2 a lane holds two contiguous 8-half runs: K = hs*8+[0,8) and hs*8+16+[0,8).
__device__ __forceinline__ v16h load_a_frag(const _Float16* rowbase, int hs) {
  const _Float16* p = rowbase + hs * 8;
  return cat8(*(const v8h*)p, *(const v8h*)(p + 16));
}

// B fragment 32x16 f16 from column-major (transposed) weights Wt[n][k]:
// lane holds two 8-half K runs for its column n: K = hs*8+[0,8) and hs*8+16+[0,8).
__device__ __forceinline__ v16h load_b_frag(const _Float16* baseT, int ldk, int n, int hs) {
  const _Float16* p = baseT + n * ldk + hs * 8;
  return cat8(*(const v8h*)p, *(const v8h*)(p + 16));
}

__device__ __forceinline__ v8f wmma_f16(v16h a, v16h b, v8f c) {
  return __builtin_amdgcn_wmma_f32_16x16x32_f16(false, a, false, b, (short)0, c, false, false);
}

__global__ __launch_bounds__(128) void gat_fused_kernel(
    const float* __restrict__ x,  const float* __restrict__ adj,
    const float* __restrict__ Wp, const float* __restrict__ bp,
    const float* __restrict__ W1, const float* __restrict__ a1v,
    const float* __restrict__ g1, const float* __restrict__ b1v,
    const float* __restrict__ W2, const float* __restrict__ a2v,
    const float* __restrict__ g2, const float* __restrict__ b2v,
    float* __restrict__ out)
{
  extern __shared__ char smem[];
  _Float16* wbuf   = (_Float16*)(smem + OFF_W);
  _Float16* hbufA  = (_Float16*)(smem + OFF_H);
  _Float16* whbufA = (_Float16*)(smem + OFF_WH);
  float* fsrcA = (float*)(smem + OFF_FS);
  float* fdstA = (float*)(smem + OFF_FD);
  float* avec  = (float*)(smem + OFF_AV);
  float* gvec  = (float*)(smem + OFF_G);
  float* bvec  = (float*)(smem + OFF_B);
  float* adjs  = (float*)(smem + OFF_ADJ);

  const int tid  = threadIdx.x;
  const int lane = tid & 31;
  const int wv   = tid >> 5;
  const int ln16 = lane & 15;     // A row / B,C column within tile
  const int hs   = lane >> 4;     // half-wave select
  const int tokBase = blockIdx.x * TOK_PER_BLK + wv * TOK_PER_WAVE;

  _Float16* myh  = hbufA  + wv * (TOK_PER_WAVE * NN * HH);  // [112][128] == [16][896]
  _Float16* mywh = whbufA + wv * (TOK_PER_WAVE * NN * HH);
  float* fsrc = fsrcA + wv * ROWS;
  float* fdst = fdstA + wv * ROWS;

  // ---------- stage Wp (transposed, f16) + adj ----------
  for (int idx = tid; idx < FIN * NN * HH; idx += blockDim.x) {
    int k = idx / (NN * HH), n = idx % (NN * HH);
    wbuf[n * FIN + k] = (_Float16)Wp[idx];
  }
  for (int idx = tid; idx < NN * NN; idx += blockDim.x) adjs[idx] = adj[idx];
  __syncthreads();

  // ---------- projection: h = x @ Wp + bp  (K = 32, one WMMA step) ----------
  {
    // A from global f32: two contiguous 8-float runs per lane (32B each)
    const float* xr = x + (tokBase + ln16) * FIN + hs * 8;
    v8h lo, hi;
#pragma unroll
    for (int t = 0; t < 8; ++t) {
      lo[t] = (_Float16)xr[t];
      hi[t] = (_Float16)xr[16 + t];
    }
    v16h af = cat8(lo, hi);
    for (int nc = 0; nc < (NN * HH) / 16; ++nc) {          // 56 column tiles
      v16h bf = load_b_frag(wbuf + nc * 16 * FIN, FIN, ln16, hs);
      v8f acc = {};
      acc = wmma_f16(af, bf, acc);
      float bias = bp[nc * 16 + ln16];
#pragma unroll
      for (int r = 0; r < 8; ++r)
        myh[(hs * 8 + r) * (NN * HH) + nc * 16 + ln16] = (_Float16)(acc[r] + bias);
    }
  }
  __syncthreads();

  const float scale = 0.08838834764831845f;  // 128^-0.5

  for (int layer = 0; layer < 2; ++layer) {
    const float* W  = layer ? W2  : W1;
    const float* av = layer ? a2v : a1v;
    const float* gg = layer ? g2  : g1;
    const float* bb = layer ? b2v : b1v;

    // ---- stage W^T (f16), a, gamma, beta ----
    for (int idx = tid; idx < HH * HH; idx += blockDim.x) {
      int k = idx >> 7, n = idx & 127;
      wbuf[n * HH + k] = (_Float16)W[idx];
    }
    for (int idx = tid; idx < 2 * HH; idx += blockDim.x) avec[idx] = av[idx];
    for (int idx = tid; idx < HH; idx += blockDim.x) { gvec[idx] = gg[idx]; bvec[idx] = bb[idx]; }
    __syncthreads();

    // ---- Wh = h @ W : rows 112, cols 128, K = 128 (4 WMMA steps) ----
    for (int rt = 0; rt < 7; ++rt) {
      const _Float16* abase = myh + (rt * 16 + ln16) * HH;
      v16h af0 = load_a_frag(abase +  0, hs);
      v16h af1 = load_a_frag(abase + 32, hs);
      v16h af2 = load_a_frag(abase + 64, hs);
      v16h af3 = load_a_frag(abase + 96, hs);
      for (int nc = 0; nc < 8; ++nc) {
        const _Float16* wb = wbuf + nc * 16 * HH;
        v8f acc = {};
        acc = wmma_f16(af0, load_b_frag(wb + 0,  HH, ln16, hs), acc);
        acc = wmma_f16(af1, load_b_frag(wb + 32, HH, ln16, hs), acc);
        acc = wmma_f16(af2, load_b_frag(wb + 64, HH, ln16, hs), acc);
        acc = wmma_f16(af3, load_b_frag(wb + 96, HH, ln16, hs), acc);
#pragma unroll
        for (int r = 0; r < 8; ++r)
          mywh[(rt * 16 + hs * 8 + r) * HH + nc * 16 + ln16] = (_Float16)acc[r];
      }
    }
    __syncthreads();

    // ---- f_src = Wh . a[:H], f_dst = Wh . a[H:], vectorized LDS reads ----
    for (int r = lane; r < ROWS; r += 32) {
      float s1 = 0.f, s2 = 0.f;
      for (int hc = 0; hc < HH; hc += 8) {
        v8h w = *(const v8h*)(mywh + r * HH + hc);
#pragma unroll
        for (int t = 0; t < 8; ++t) {
          float f = (float)w[t];
          s1 += f * avec[hc + t];
          s2 += f * avec[HH + hc + t];
        }
      }
      fsrc[r] = s1; fdst[r] = s2;
    }
    __syncthreads();

    // ---- attention softmax + h' = att@Wh + layernorm + residual ----
    for (int r = lane; r < ROWS; r += 32) {
      int tok = r / NN, i = r % NN;
      float fs = fsrc[r];
      float att[NN];
      float mx = -3.4e38f;
#pragma unroll
      for (int j = 0; j < NN; ++j) {
        float e = fs + fdst[tok * NN + j];
        e = (e > 0.f ? e : 0.2f * e) * scale;           // leaky relu then scale
        if (!(adjs[i * NN + j] > 0.f)) e = -9.0e15f;    // mask
        att[j] = e;
        mx = fmaxf(mx, e);
      }
      float den = 0.f;
#pragma unroll
      for (int j = 0; j < NN; ++j) { att[j] = __expf(att[j] - mx); den += att[j]; }
      float inv = 1.f / den;
#pragma unroll
      for (int j = 0; j < NN; ++j) att[j] *= inv;

      const _Float16* wrow = mywh + tok * NN * HH;
      // pass 1: mean / var of h' (recompute avoids a 57KB/wave scratch buffer)
      float sum = 0.f, sq = 0.f;
      for (int hc = 0; hc < HH; hc += 8) {
        v8h wj[NN];
#pragma unroll
        for (int j = 0; j < NN; ++j) wj[j] = *(const v8h*)(wrow + j * HH + hc);
#pragma unroll
        for (int t = 0; t < 8; ++t) {
          float hp = 0.f;
#pragma unroll
          for (int j = 0; j < NN; ++j) hp += att[j] * (float)wj[j][t];
          sum += hp; sq += hp * hp;
        }
      }
      float mean = sum * (1.f / HH);
      float var  = sq  * (1.f / HH) - mean * mean;
      float rstd = rsqrtf(var + 1e-5f);
      // pass 2: normalize + residual, write new h in place (vector LDS I/O)
      for (int hc = 0; hc < HH; hc += 8) {
        v8h wj[NN];
#pragma unroll
        for (int j = 0; j < NN; ++j) wj[j] = *(const v8h*)(wrow + j * HH + hc);
        v8h res = *(const v8h*)(myh + r * HH + hc);
        v8h o;
#pragma unroll
        for (int t = 0; t < 8; ++t) {
          float hp = 0.f;
#pragma unroll
          for (int j = 0; j < NN; ++j) hp += att[j] * (float)wj[j][t];
          o[t] = (_Float16)((hp - mean) * rstd * gvec[hc + t] + bvec[hc + t] + (float)res[t]);
        }
        *(v8h*)(myh + r * HH + hc) = o;
      }
    }
    __syncthreads();
  }

  // ---------- output: mean over 7 nodes, fp32, vectorized & coalesced ----------
  for (int c = tid; c < (TOK_PER_BLK * HH) / 8; c += blockDim.x) {  // 1024 chunks
    int tok = c >> 4;
    int hc  = (c & 15) << 3;
    const _Float16* hb = hbufA + (tok >> 4) * (TOK_PER_WAVE * NN * HH)
                               + (tok & 15) * (NN * HH) + hc;
    float s[8] = {0.f, 0.f, 0.f, 0.f, 0.f, 0.f, 0.f, 0.f};
#pragma unroll
    for (int j = 0; j < NN; ++j) {
      v8h w = *(const v8h*)(hb + j * HH);
#pragma unroll
      for (int t = 0; t < 8; ++t) s[t] += (float)w[t];
    }
    float* op = out + (blockIdx.x * TOK_PER_BLK + tok) * HH + hc;
    const float inv7 = 1.f / 7.f;
    float4 o0 = make_float4(s[0] * inv7, s[1] * inv7, s[2] * inv7, s[3] * inv7);
    float4 o1 = make_float4(s[4] * inv7, s[5] * inv7, s[6] * inv7, s[7] * inv7);
    *(float4*)op       = o0;
    *(float4*)(op + 4) = o1;
  }
}

extern "C" void kernel_launch(void* const* d_in, const int* in_sizes, int n_in,
                              void* d_out, int out_size, void* d_ws, size_t ws_size,
                              hipStream_t stream) {
  (void)in_sizes; (void)n_in; (void)d_ws; (void)ws_size; (void)out_size;
  const float* x   = (const float*)d_in[0];
  const float* adj = (const float*)d_in[1];
  const float* Wp  = (const float*)d_in[2];
  const float* bp  = (const float*)d_in[3];
  const float* W1  = (const float*)d_in[4];
  const float* a1v = (const float*)d_in[5];
  const float* g1  = (const float*)d_in[6];
  const float* b1v = (const float*)d_in[7];
  const float* W2  = (const float*)d_in[8];
  const float* a2v = (const float*)d_in[9];
  const float* g2  = (const float*)d_in[10];
  const float* b2v = (const float*)d_in[11];
  float* out = (float*)d_out;

  // allow >64KB dynamic LDS (WGP supports 320KB); harmless if a no-op
  (void)hipFuncSetAttribute((const void*)gat_fused_kernel,
                            hipFuncAttributeMaxDynamicSharedMemorySize, SMEM_BYTES);

  dim3 grid(BT / TOK_PER_BLK);   // 256 blocks
  dim3 block(128);               // 4 waves
  hipLaunchKernelGGL(gat_fused_kernel, grid, block, SMEM_BYTES, stream,
                     x, adj, Wp, bp, W1, a1v, g1, b1v, W2, a2v, g2, b2v, out);
}
